// lstmNN_68925635166324
// MI455X (gfx1250) — matrix operand order repaired and checked by reading, not compiled, not run on hardware
//
#include <hip/hip_runtime.h>
#include <hip/hip_bf16.h>

// ---------------------------------------------------------------------------
// LSTM NN on MI455X (gfx1250):  B=1024, S=512, I=64, H=128, O=1
//   feas collapse:  r = x @ (W1@W0)^T + (W1@b0 + b1);  xr = relu(r + x)
//   gates:          xg = xr @ W_ih^T + (b_ih + b_hh)     (precomputed, f16)
//   scan:           per-16-batch-rows workgroup runs all 512 steps privately
//
// xg staging layout (f16, "C-fragment packed"):
//   xg[(s*64+bt)*8192 + ct*256 + lane*8 + v]
// so each lane's 8 accumulator-init values are one 16-byte global_load_b128,
// and a wave covers a contiguous 512 B block (fully coalesced).
// ---------------------------------------------------------------------------

typedef _Float16 h8   __attribute__((ext_vector_type(8)));
typedef _Float16 h16  __attribute__((ext_vector_type(16)));
typedef float    f8   __attribute__((ext_vector_type(8)));

constexpr int kB  = 1024;
constexpr int kS  = 512;
constexpr int kI  = 64;
constexpr int kH  = 128;
constexpr int kBT = kB / 16;      // 64 batch tiles of 16 rows

constexpr size_t kXgBytes = (size_t)kS * kB * 512 * sizeof(_Float16); // 512 MB
constexpr size_t kWcOff   = kXgBytes;            // 64*64 f16
constexpr size_t kBcOff   = kXgBytes + 8192;     // 64 f32

static __device__ __forceinline__ h16 cat8(h8 lo, h8 hi) {
  return __builtin_shufflevector(lo, hi, 0,1,2,3,4,5,6,7,8,9,10,11,12,13,14,15);
}

// 16 contiguous f32 -> h16 (B-fragment: W[n][k..k+15])
static __device__ __forceinline__ h16 cvt16(const float* p) {
  const float4 a = *(const float4*)(p + 0);
  const float4 b = *(const float4*)(p + 4);
  const float4 c = *(const float4*)(p + 8);
  const float4 d = *(const float4*)(p + 12);
  h16 r;
  r[0]=(_Float16)a.x; r[1]=(_Float16)a.y; r[2]=(_Float16)a.z; r[3]=(_Float16)a.w;
  r[4]=(_Float16)b.x; r[5]=(_Float16)b.y; r[6]=(_Float16)b.z; r[7]=(_Float16)b.w;
  r[8]=(_Float16)c.x; r[9]=(_Float16)c.y; r[10]=(_Float16)c.z; r[11]=(_Float16)c.w;
  r[12]=(_Float16)d.x; r[13]=(_Float16)d.y; r[14]=(_Float16)d.z; r[15]=(_Float16)d.w;
  return r;
}

// two chunks of 8 contiguous f32 -> h16 (A-fragment: row[k..k+7], row[k+8..k+15])
static __device__ __forceinline__ h16 cvt8x2(const float* p0, const float* p1) {
  const float4 a = *(const float4*)(p0 + 0);
  const float4 b = *(const float4*)(p0 + 4);
  const float4 c = *(const float4*)(p1 + 0);
  const float4 d = *(const float4*)(p1 + 4);
  h16 r;
  r[0]=(_Float16)a.x; r[1]=(_Float16)a.y; r[2]=(_Float16)a.z; r[3]=(_Float16)a.w;
  r[4]=(_Float16)b.x; r[5]=(_Float16)b.y; r[6]=(_Float16)b.z; r[7]=(_Float16)b.w;
  r[8]=(_Float16)c.x; r[9]=(_Float16)c.y; r[10]=(_Float16)c.z; r[11]=(_Float16)c.w;
  r[12]=(_Float16)d.x; r[13]=(_Float16)d.y; r[14]=(_Float16)d.z; r[15]=(_Float16)d.w;
  return r;
}

static __device__ __forceinline__ f8 wmma_f16(h16 a, h16 b, f8 c) {
  return __builtin_amdgcn_wmma_f32_16x16x32_f16(false, a, false, b, (short)0, c,
                                                false, false);
}

static __device__ __forceinline__ float sigm(float x) {
  return 1.0f / (1.0f + __expf(-x));
}

// ---------------------------------------------------------------------------
// Kernel 1: Wc = W1 @ W0  (64x64, f16)  and  bc = W1 @ b0 + b1 (f32)
// ---------------------------------------------------------------------------
__global__ __launch_bounds__(256) void k_collapse(const float* __restrict__ W0,
                                                  const float* __restrict__ b0,
                                                  const float* __restrict__ W1,
                                                  const float* __restrict__ b1,
                                                  _Float16* __restrict__ Wc,
                                                  float* __restrict__ bc) {
  const int t = threadIdx.x;
  for (int idx = t; idx < kI * kI; idx += 256) {
    const int i = idx >> 6, k = idx & 63;
    float acc = 0.f;
#pragma unroll 8
    for (int j = 0; j < 4 * kI; ++j) acc += W1[i * 256 + j] * W0[j * 64 + k];
    Wc[i * 64 + k] = (_Float16)acc;
  }
  if (t < kI) {
    float a = b1[t];
#pragma unroll 8
    for (int j = 0; j < 4 * kI; ++j) a += W1[t * 256 + j] * b0[j];
    bc[t] = a;
  }
}

// ---------------------------------------------------------------------------
// Kernel 2: xg = relu(x@Wc^T + bc + x) @ W_ih^T + (b_ih+b_hh)
// One 128-thread WG per (s, batch-tile); fragment-packed f16 output.
// ---------------------------------------------------------------------------
__global__ __launch_bounds__(128) void k_feed(const float* __restrict__ x,
                                              const _Float16* __restrict__ Wc,
                                              const float* __restrict__ bc,
                                              const float* __restrict__ W_ih,
                                              const float* __restrict__ b_ih,
                                              const float* __restrict__ b_hh,
                                              _Float16* __restrict__ xg) {
  __shared__ __align__(16) _Float16 xr[16 * kI];  // relu(r + x) tile, f16

  const int sb   = blockIdx.x;          // s*64 + bt
  const int s    = sb >> 6;
  const int bt   = sb & 63;
  const int w    = threadIdx.x >> 5;    // 4 waves
  const int lane = threadIdx.x & 31;
  const int l16  = lane & 15;
  const int lhi  = lane >> 4;

  // ---- phase A: r = x @ Wc^T, wave w -> columns 16w..16w+15 ----------------
  const size_t xrow = ((size_t)(bt * 16 + l16) * kS + s) * kI;  // this lane's A row
  h16 A[2];
#pragma unroll
  for (int ks = 0; ks < 2; ++ks) {
    const int k = ks * 32 + lhi * 16;
    A[ks] = cvt8x2(x + xrow + k, x + xrow + k + 8);
  }
  h16 Bw[2];
  const _Float16* wcr = Wc + (w * 16 + l16) * kI;
#pragma unroll
  for (int ks = 0; ks < 2; ++ks) {
    const int k = ks * 32 + lhi * 16;
    Bw[ks] = cat8(*(const h8*)(wcr + k), *(const h8*)(wcr + k + 8));
  }
  f8 acc = {};
  acc = wmma_f16(A[0], Bw[0], acc);
  acc = wmma_f16(A[1], Bw[1], acc);

  const int n0 = w * 16 + l16;
  const float bcn = bc[n0];
#pragma unroll
  for (int v = 0; v < 8; ++v) {
    const int row = v + lhi * 8;
    const float xv = x[((size_t)(bt * 16 + row) * kS + s) * kI + n0];
    float rv = acc[v] + bcn + xv;
    rv = rv > 0.f ? rv : 0.f;
    xr[row * kI + n0] = (_Float16)rv;
  }
  __syncthreads();

  // ---- phase B: xg = xr @ W_ih^T, wave w -> column tiles {w, w+4, ...} -----
  h16 A2[2];
#pragma unroll
  for (int ks = 0; ks < 2; ++ks) {
    const int k = ks * 32 + lhi * 16;
    A2[ks] = cat8(*(const h8*)&xr[l16 * kI + k],
                  *(const h8*)&xr[l16 * kI + k + 8]);
  }
  _Float16* outb = xg + (size_t)sb * (32 * 256);
#pragma unroll
  for (int cc = 0; cc < 8; ++cc) {
    const int ct = cc * 4 + w;
    const int n  = ct * 16 + l16;
    const float* wr = W_ih + (size_t)n * kI;
    h16 Bf0 = cvt16(wr + 0 + lhi * 16);        // ks=0: k = lhi*16
    h16 Bf1 = cvt16(wr + 32 + lhi * 16);       // ks=1: k = 32 + lhi*16
    const float bg = b_ih[n] + b_hh[n];
    f8 a;
#pragma unroll
    for (int v = 0; v < 8; ++v) a[v] = bg;
    a = wmma_f16(A2[0], Bf0, a);
    a = wmma_f16(A2[1], Bf1, a);
    // fragment-packed store: one b128 per lane per tile
    h8 pk;
#pragma unroll
    for (int v = 0; v < 8; ++v) pk[v] = (_Float16)a[v];
    *(h8*)(outb + ct * 256 + lane * 8) = pk;
  }
}

// ---------------------------------------------------------------------------
// Kernel 3: recurrent scan.  One 256-thread WG per 16 batch rows; wave w owns
// hidden units 16w..16w+15 for ALL four gates (ct = w, 8+w, 16+w, 24+w), so
// cell state and gate math never leave the wave.  W_hh lives in registers as
// 16 f16 B-fragments; h lives in 4 KB of LDS (f16), rewritten each step.
// Per step/wave: 4 global_load_b128 (xg), 8 ds_load_b128 (h), 16 WMMA.
// Epilogue fuses  out = h_T @ Wf^T + bf.
// ---------------------------------------------------------------------------
__global__ __launch_bounds__(256, 1) void k_scan(const float* __restrict__ W_hh,
                                                 const float* __restrict__ Wf,
                                                 const float* __restrict__ bf,
                                                 const _Float16* __restrict__ xg,
                                                 float* __restrict__ out) {
  __shared__ __align__(16) _Float16 hlds[16 * kH];  // h_prev tile (16 x 128)

  const int w    = threadIdx.x >> 5;   // 8 waves
  const int lane = threadIdx.x & 31;
  const int l16  = lane & 15;
  const int lhi  = lane >> 4;
  const int bt   = blockIdx.x;

  // W_hh -> f16 B fragments in registers: Bf[q][ks], q = gate, K = 128
  h16 Bf[16];
#pragma unroll
  for (int q = 0; q < 4; ++q) {
    const int n = q * kH + w * 16 + l16;          // row of W_hh [512][128]
    const float* wr = W_hh + (size_t)n * kH;
#pragma unroll
    for (int ks = 0; ks < 4; ++ks)
      Bf[q * 4 + ks] = cvt16(wr + ks * 32 + lhi * 16);
  }

  // zero h, c
  for (int i = threadIdx.x; i < 16 * kH; i += 256) hlds[i] = (_Float16)0.f;
  f8 cst = {};
  __syncthreads();

  const int hj = w * 16 + l16;                    // hidden unit this lane owns

  for (int s = 0; s < kS; ++s) {
    // A fragments from h_prev (LDS)
    h16 A[4];
#pragma unroll
    for (int ks = 0; ks < 4; ++ks) {
      const int k = ks * 32 + lhi * 16;
      A[ks] = cat8(*(const h8*)&hlds[l16 * kH + k],
                   *(const h8*)&hlds[l16 * kH + k + 8]);
    }
    __syncthreads();   // all reads of h done before anyone rewrites it

    const _Float16* xgs = xg + ((size_t)s * kBT + bt) * (32 * 256);
    if (s + 1 < kS)    // pull next step's fragment block toward L2/L0
      __builtin_prefetch(xg + ((size_t)(s + 1) * kBT + bt) * (32 * 256)
                            + w * 256 + lane * 8, 0, 1);

    f8 acc[4];
#pragma unroll
    for (int q = 0; q < 4; ++q) {
      // one b128 load per lane: this lane's 8 accumulator-init values
      const h8 xv = *(const h8*)(xgs + (q * 8 + w) * 256 + lane * 8);
      f8 a;
#pragma unroll
      for (int v = 0; v < 8; ++v) a[v] = (float)xv[v];
#pragma unroll
      for (int ks = 0; ks < 4; ++ks) a = wmma_f16(A[ks], Bf[q * 4 + ks], a);
      acc[q] = a;
    }

    // gates: i, f, g, o  ->  c, h
#pragma unroll
    for (int v = 0; v < 8; ++v) {
      const float gi = acc[0][v], gf = acc[1][v], gg = acc[2][v], go = acc[3][v];
      const float cn = sigm(gf) * cst[v] + sigm(gi) * tanhf(gg);
      const float hn = sigm(go) * tanhf(cn);
      cst[v] = cn;
      hlds[(v + lhi * 8) * kH + hj] = (_Float16)hn;
    }
    __syncthreads();
  }

  // epilogue: out[b] = h_T[b] . Wf + bf   (h_T already in LDS)
  if (threadIdx.x < 16) {
    const int r = threadIdx.x;
    float acc = bf[0];
#pragma unroll 8
    for (int j = 0; j < kH; ++j) acc += (float)hlds[r * kH + j] * Wf[j];
    out[bt * 16 + r] = acc;
  }
}

// ---------------------------------------------------------------------------
extern "C" void kernel_launch(void* const* d_in, const int* in_sizes, int n_in,
                              void* d_out, int out_size, void* d_ws, size_t ws_size,
                              hipStream_t stream) {
  const float* x    = (const float*)d_in[0];
  const float* W0   = (const float*)d_in[1];
  const float* b0   = (const float*)d_in[2];
  const float* W1   = (const float*)d_in[3];
  const float* b1   = (const float*)d_in[4];
  const float* W_ih = (const float*)d_in[5];
  const float* W_hh = (const float*)d_in[6];
  const float* b_ih = (const float*)d_in[7];
  const float* b_hh = (const float*)d_in[8];
  const float* Wf   = (const float*)d_in[9];
  const float* bf   = (const float*)d_in[10];
  float* out = (float*)d_out;

  char* ws = (char*)d_ws;
  _Float16* xgws = (_Float16*)ws;
  _Float16* Wc   = (_Float16*)(ws + kWcOff);
  float*    bc   = (float*)(ws + kBcOff);

  k_collapse<<<1, 256, 0, stream>>>(W0, b0, W1, b1, Wc, bc);
  k_feed<<<kS * kBT, 128, 0, stream>>>(x, Wc, bc, W_ih, b_ih, b_hh, xgws);
  k_scan<<<kBT, 256, 0, stream>>>(W_hh, Wf, bf, xgws, out);
}
